// DecoderStep_71966472012229
// MI455X (gfx1250) — matrix operand order repaired and checked by reading, not compiled, not run on hardware
//
#include <hip/hip_runtime.h>
#include <hip/hip_bf16.h>
#include <stdint.h>

// ---------------- problem constants ----------------
#define B     2048
#define S     512
#define H     50
#define E     50
#define V     32000
#define KIN   150          // E + 2H classifier input width
#define KPAD  160          // padded to multiple of 32 for WMMA f16 K-steps
#define MT    4            // 16-row M tiles per classifier block (64 rows)

typedef __attribute__((ext_vector_type(16))) _Float16 v16h;
typedef __attribute__((ext_vector_type(8)))  _Float16 v8h;
typedef __attribute__((ext_vector_type(8)))  float    v8f;
typedef __attribute__((ext_vector_type(4)))  unsigned int u32x4;
typedef __attribute__((ext_vector_type(8)))  int      i32x8;
typedef __attribute__((ext_vector_type(4)))  int      i32x4;

// ---------------- workspace layout (bytes) ----------------
// ctx     : B*H f32                =   409,600
// cls_in  : B*KPAD f16             =   655,360
// W_cls_h : V*KPAD f16             = 10,240,000   (32B aligned)
#define WS_CTX     ((size_t)0)
#define WS_CLSIN   ((size_t)409600)
#define WS_WCLSH   ((size_t)1064960)

__device__ __forceinline__ float sigmoidf_(float x) {
    return 1.0f / (1.0f + __expf(-x));
}
// tanh via the v_exp_f32 TRANS unit: 1 - 2/(e^(2x)+1); correct limits at +-inf
__device__ __forceinline__ float tanh_fast(float x) {
    return 1.0f - 2.0f / (__expf(2.0f * x) + 1.0f);
}

// ------------- Tensor Data Mover: 2D tile Global -> LDS -------------
// dsz code: 0=1B, 1=2B, 2=4B, 3=8B. dims/strides in elements.
__device__ __forceinline__ void tdm_load_2d(const void* g, void* l,
                                            uint32_t dim0, uint32_t dim1,
                                            uint64_t stride0, uint32_t dsz) {
    uint64_t ga = (uint64_t)(uintptr_t)g;
    uint32_t la = (uint32_t)(uintptr_t)l;          // LDS offset = flat addr[31:0]
    // D# group 0: count=1 | lds_addr | global_addr[56:0] | type=2
    u32x4 g0 = { 1u,
                 la,
                 (uint32_t)ga,
                 ((uint32_t)(ga >> 32) & 0x01FFFFFFu) | (2u << 30) };
    uint64_t s1 = (uint64_t)dim0 * dim1;           // dim1 stride (unused for 2D walk)
    // D# group 1: mask=0, data_size, tensor_dim0/1, tile_dim0/1/2, dim strides
    i32x8 g1 = { (int)(dsz << 16),
                 (int)((dim0 & 0xFFFFu) << 16),
                 (int)((dim0 >> 16) | ((dim1 & 0xFFFFu) << 16)),
                 (int)((dim1 >> 16) | (dim0 << 16)),          // tile_dim0 = dim0
                 (int)(dim1 & 0xFFFFu),                        // tile_dim1 = dim1
                 (int)(stride0 & 0xFFFFFFFFu),
                 (int)((stride0 >> 32) | ((s1 & 0xFFFFu) << 16)),
                 (int)(s1 >> 16) };
    i32x4 z4 = {0, 0, 0, 0};
#if __clang_major__ >= 23
    i32x8 z8 = {0, 0, 0, 0, 0, 0, 0, 0};
    __builtin_amdgcn_tensor_load_to_lds(g0, g1, z4, z4, z8, 0);
#else
    __builtin_amdgcn_tensor_load_to_lds(g0, g1, z4, z4, 0);
#endif
}

// ---------------- K0: W_cls f32 -> f16, pad K 150->160 ----------------
__global__ void k_cvt_wcls(const float* __restrict__ Wcls, _Float16* __restrict__ Wch) {
    int gid = blockIdx.x * blockDim.x + threadIdx.x;
    if (gid >= V * KPAD) return;
    int n = gid / KPAD, k = gid - n * KPAD;
    Wch[gid] = (k < KIN) ? (_Float16)Wcls[(size_t)n * KIN + k] : (_Float16)0.0f;
}

// ---------------- K1: fused attention (one block per batch row) ----------------
// TDM stages enc[:, b, :] (512 x 50 f32, row stride B*H) into LDS once; then
// proj_prev, scores+tanh, softmax over S, and ctx are all computed from LDS.
__global__ void k_attention(const float* __restrict__ enc, const float* __restrict__ h,
                            const float* __restrict__ Wp, const float* __restrict__ We,
                            const float* __restrict__ v, float* __restrict__ ctx) {
    extern __shared__ float encS[];          // S*H f32 = 100 KB (dynamic)
    __shared__ float WeS[H * H];
    __shared__ float vS[H];
    __shared__ float pp[H];
    __shared__ float sc[S];
    __shared__ float red[256];

    int b = blockIdx.x, t = threadIdx.x;

    if ((t >> 5) == 0) {                     // wave 0 drives the DMA
        tdm_load_2d(enc + (size_t)b * H, encS, /*dim0=*/H, /*dim1=*/S,
                    /*stride0=*/(uint64_t)B * H, /*dsz f32=*/2u);
        __builtin_amdgcn_s_wait_tensorcnt(0);
    }
    for (int i = t; i < H * H; i += 256) WeS[i] = We[i];   // overlaps with TDM
    if (t < H) vS[t] = v[t];
    __syncthreads();

    // proj_prev[k] = sum_j h[b,j] * Wp[k,j]
    if (t < H) {
        float acc = 0.f;
#pragma unroll
        for (int j = 0; j < H; ++j) acc += h[b * H + j] * Wp[t * H + j];
        pp[t] = acc;
    }
    __syncthreads();

    // scores[s] = sum_k tanh(pp[k] + enc_row . We[k,:]) * v[k]
    for (int s = t; s < S; s += 256) {
        float e[H];
#pragma unroll
        for (int j = 0; j < H; ++j) e[j] = encS[s * H + j];
        float scv = 0.f;
        for (int k = 0; k < H; ++k) {
            float a = pp[k];
#pragma unroll
            for (int j = 0; j < H; ++j) a += e[j] * WeS[k * H + j];
            scv += tanh_fast(a) * vS[k];
        }
        sc[s] = scv;
    }
    __syncthreads();

    // softmax over S in LDS
    float e0 = sc[t], e1 = sc[t + 256];
    red[t] = fmaxf(e0, e1);
    __syncthreads();
    for (int off = 128; off > 0; off >>= 1) {
        if (t < off) red[t] = fmaxf(red[t], red[t + off]);
        __syncthreads();
    }
    float mx = red[0];
    __syncthreads();
    float x0 = __expf(e0 - mx), x1 = __expf(e1 - mx);
    red[t] = x0 + x1;
    __syncthreads();
    for (int off = 128; off > 0; off >>= 1) {
        if (t < off) red[t] += red[t + off];
        __syncthreads();
    }
    float inv = 1.0f / red[0];
    sc[t] = x0 * inv;
    sc[t + 256] = x1 * inv;
    __syncthreads();

    // ctx[b,h] = sum_s attn[s] * enc[s,h]   (from LDS; lanes hit distinct banks)
    if (t < H) {
        float acc = 0.f;
        for (int s = 0; s < S; ++s) acc += sc[s] * encS[s * H + t];
        ctx[b * H + t] = acc;
    }
}

// ---------------- K2: LSTM cell + build f16 cls_in ----------------
__global__ void k_lstm(const int* __restrict__ ids, const float* __restrict__ h,
                       const float* __restrict__ c, const float* __restrict__ emb,
                       const float* __restrict__ ctx,
                       const float* __restrict__ Wih, const float* __restrict__ Whh,
                       const float* __restrict__ bih, const float* __restrict__ bhh,
                       float* __restrict__ h_out, float* __restrict__ c_out,
                       _Float16* __restrict__ cls_in) {
    __shared__ float xs[E + H];   // [emb(50) | ctx(50)]
    __shared__ float hs[H];
    __shared__ float gs[4 * H];
    int b = blockIdx.x, t = threadIdx.x;
    if (t < E)            xs[t] = emb[(size_t)ids[b] * E + t];
    else if (t < E + H)   xs[t] = ctx[b * H + (t - E)];
    if (t < H)            hs[t] = h[b * H + t];
    __syncthreads();

    if (t < 4 * H) {
        float g = bih[t] + bhh[t];
#pragma unroll 10
        for (int k = 0; k < E + H; ++k) g += Wih[t * (E + H) + k] * xs[k];
#pragma unroll 10
        for (int k = 0; k < H; ++k)     g += Whh[t * H + k] * hs[k];
        gs[t] = g;
    }
    __syncthreads();

    if (t < H) {
        float ig = sigmoidf_(gs[t]);
        float fg = sigmoidf_(gs[H + t]);
        float gg = tanh_fast(gs[2 * H + t]);
        float og = sigmoidf_(gs[3 * H + t]);
        float cn = fg * c[b * H + t] + ig * gg;
        float hn = og * tanh_fast(cn);
        h_out[b * H + t] = hn;
        c_out[b * H + t] = cn;
        _Float16* row = cls_in + (size_t)b * KPAD;
        row[t]          = (_Float16)xs[t];        // prev_emb
        row[E + t]      = (_Float16)hn;           // h_new
        row[E + H + t]  = (_Float16)xs[E + t];    // ctx
        if (t < KPAD - KIN) row[KIN + t] = (_Float16)0.0f;  // zero pad
    }
}

// ---------------- K3: classifier GEMM via WMMA f16 -> f32 ----------------
// grid = (V/128, B/64); block = 256 (8 waves). Block tile = 64 rows x 128 cols.
// A tile (64x160 f16) TDM-loaded to LDS; each B fetch feeds MT=4 WMMAs.
__global__ void k_classifier(const _Float16* __restrict__ clsin,
                             const _Float16* __restrict__ Wch,
                             const float* __restrict__ bcls,
                             float* __restrict__ logits) {
    __shared__ __attribute__((aligned(32))) _Float16 Asm[MT * 16 * KPAD];  // 20 KB

    int b0 = blockIdx.y * (MT * 16);
    if ((threadIdx.x >> 5) == 0) {
        tdm_load_2d(clsin + (size_t)b0 * KPAD, Asm, /*dim0=*/KPAD, /*dim1=*/MT * 16,
                    /*stride0=*/KPAD, /*dsz f16=*/1u);
        __builtin_amdgcn_s_wait_tensorcnt(0);
    }
    __syncthreads();

    int lane = threadIdx.x & 31;
    int w    = threadIdx.x >> 5;
    int n0   = blockIdx.x * 128 + w * 16;
    int nrow = n0 + (lane & 15);
    int kb2  = (lane < 16) ? 0 : 16;   // B operand: high lanes hold K=16..31 of step
    int aoff = (lane < 16) ? 0 : 16;   // A operand: high lanes hold K=16..31 of step
    int m    = lane & 15;

    const _Float16* Brow = Wch + (size_t)nrow * KPAD + kb2;
    v8f acc[MT];
#pragma unroll
    for (int tmt = 0; tmt < MT; ++tmt) acc[tmt] = (v8f){};

#pragma unroll
    for (int ks = 0; ks < KPAD / 32; ++ks) {
        if (ks + 1 < KPAD / 32)
            __builtin_prefetch(Brow + (ks + 1) * 32, 0, 1);   // global_prefetch_b8
        v16h bmat = *(const v16h*)(Brow + ks * 32);
#pragma unroll
        for (int tmt = 0; tmt < MT; ++tmt) {
            const _Float16* arow = Asm + (tmt * 16 + m) * KPAD + ks * 32 + aoff;
            v8h alo = *(const v8h*)(arow);
            v8h ahi = *(const v8h*)(arow + 8);
            v16h amat;
#pragma unroll
            for (int i = 0; i < 8; ++i) { amat[i] = alo[i]; amat[i + 8] = ahi[i]; }
            acc[tmt] = __builtin_amdgcn_wmma_f32_16x16x32_f16(
                false, amat, false, bmat, (short)0, acc[tmt], false, false);
        }
    }

    int col = n0 + (lane & 15);
    float bias = bcls[col];
    int mb = (lane < 16) ? 0 : 8;   // D layout: VGPR r -> M = r (+8 for high lanes)
#pragma unroll
    for (int tmt = 0; tmt < MT; ++tmt)
#pragma unroll
        for (int r = 0; r < 8; ++r)
            logits[(size_t)(b0 + tmt * 16 + r + mb) * V + col] = acc[tmt][r] + bias;
}

// ---------------- K4: row softmax over V, online max+sum (2 passes) ----------------
__global__ void k_softmax_v(float* __restrict__ logits) {
    __shared__ float redm[256];
    __shared__ float reds[256];
    int b = blockIdx.x, t = threadIdx.x;
    float* row = logits + (size_t)b * V;
    const int PER = V / 256;  // 125

    float m = -INFINITY, s = 0.f;
    for (int i = 0; i < PER; ++i) {
        float x = row[t + i * 256];
        float nm = fmaxf(m, x);
        s = s * __expf(m - nm) + __expf(x - nm);
        m = nm;
    }
    redm[t] = m;
    reds[t] = s;
    __syncthreads();
    for (int off = 128; off > 0; off >>= 1) {
        if (t < off) {
            float m2 = redm[t + off], s2 = reds[t + off];
            float nm = fmaxf(redm[t], m2);
            reds[t] = reds[t] * __expf(redm[t] - nm) + s2 * __expf(m2 - nm);
            redm[t] = nm;
        }
        __syncthreads();
    }
    float mx = redm[0];
    float inv = 1.0f / reds[0];
    for (int i = 0; i < PER; ++i) {
        int idx = t + i * 256;
        row[idx] = __expf(row[idx] - mx) * inv;
    }
}

// ---------------- launcher ----------------
extern "C" void kernel_launch(void* const* d_in, const int* in_sizes, int n_in,
                              void* d_out, int out_size, void* d_ws, size_t ws_size,
                              hipStream_t stream) {
    const int*   ids  = (const int*)  d_in[0];
    const float* h    = (const float*)d_in[1];
    const float* c    = (const float*)d_in[2];
    const float* enc  = (const float*)d_in[3];
    const float* emb  = (const float*)d_in[4];
    const float* Wp   = (const float*)d_in[5];
    const float* We   = (const float*)d_in[6];
    const float* v    = (const float*)d_in[7];
    const float* Wih  = (const float*)d_in[8];
    const float* Whh  = (const float*)d_in[9];
    const float* bih  = (const float*)d_in[10];
    const float* bhh  = (const float*)d_in[11];
    const float* Wcls = (const float*)d_in[12];
    const float* bcls = (const float*)d_in[13];

    float* out    = (float*)d_out;
    float* h_out  = out;                 // (B,H)
    float* c_out  = out + B * H;         // (B,H)
    float* preds  = out + 2 * B * H;     // (B,V)

    char* ws = (char*)d_ws;
    float*    ctx    = (float*)(ws + WS_CTX);
    _Float16* clsin  = (_Float16*)(ws + WS_CLSIN);
    _Float16* Wch    = (_Float16*)(ws + WS_WCLSH);

    k_cvt_wcls  <<<(V * KPAD + 255) / 256, 256, 0,           stream>>>(Wcls, Wch);
    k_attention <<<B,                      256, S * H * 4,   stream>>>(enc, h, Wp, We, v, ctx);
    k_lstm      <<<B,                      256, 0,           stream>>>(ids, h, c, emb, ctx,
                                                                       Wih, Whh, bih, bhh,
                                                                       h_out, c_out, clsin);
    k_classifier<<<dim3(V / 128, B / (MT * 16)), 256, 0,     stream>>>(clsin, Wch, bcls, preds);
    k_softmax_v <<<B,                      256, 0,           stream>>>(preds);
}
